// SequenceEncoder_11355893531210
// MI455X (gfx1250) — matrix operand order repaired and checked by reading, not compiled, not run on hardware
//
#include <hip/hip_runtime.h>

// ---------------------------------------------------------------------------
// Types
// ---------------------------------------------------------------------------
typedef __attribute__((ext_vector_type(16))) __bf16 v16bf;
typedef __attribute__((ext_vector_type(8)))  float  v8f;

#define DM   128
#define KNB  32
#define NN   4096
#define BB   2
#define BNN  (BB * NN)          // 8192 nodes
#define EPSL 1e-5f

// ---------------------------------------------------------------------------
// Device helpers
// ---------------------------------------------------------------------------
__device__ __forceinline__ v8f wmma_bf16(v16bf a, v16bf b, v8f c) {
  return __builtin_amdgcn_wmma_f32_16x16x32_bf16(false, a, false, b, (short)0, c,
                                                 false, false);
}

__device__ __forceinline__ float gelu_t(float x) {
  float x3 = x * x * x;
  return 0.5f * x * (1.0f + tanhf(0.7978845608f * (x + 0.044715f * x3)));
}

// B fragment: pre-swizzled bf16 weights; one contiguous 32B run per lane.
__device__ __forceinline__ v16bf bfrag(const __bf16* wf, int NT, int kc, int nt,
                                       int lane) {
  return *(const v16bf*)(wf + (((size_t)(kc * NT + nt) * 32 + lane) << 4));
}

// A fragment from f32 global memory, row pointer already offset by kc*32.
// ISA 16-bit A layout: elem i<8 -> K = hb*8+i ; elem 8+i -> K = 16+hb*8+i.
__device__ __forceinline__ v16bf afrag_g32(const float* rowbase, int lane) {
  int hb = (lane >> 4) & 1;
  const float* p = rowbase + hb * 8;
  v16bf a;
#pragma unroll
  for (int i = 0; i < 8; ++i) {
    a[i]     = (__bf16)p[i];
    a[8 + i] = (__bf16)p[16 + i];
  }
  return a;
}

// A fragment from an LDS bf16 tile (row-major, leading dim ldw).
__device__ __forceinline__ v16bf afrag_lds(const __bf16* tile, int ldw, int kc,
                                           int lane) {
  int row = lane & 15, hb = (lane >> 4) & 1;
  const __bf16* p = tile + row * ldw + kc * 32 + hb * 8;
  v16bf a;
#pragma unroll
  for (int i = 0; i < 8; ++i) {
    a[i]     = p[i];
    a[8 + i] = p[16 + i];
  }
  return a;
}

// Store a 16x16 f32 C tile into a bf16 LDS tile with bias (+optional gelu).
__device__ __forceinline__ void store_h(__bf16* tile, int ldw, int nt, int lane,
                                        v8f c, const float* bias, bool act) {
  int col = nt * 16 + (lane & 15);
  int hb  = (lane >> 4) & 1;
  float bv = bias[col];
#pragma unroll
  for (int r = 0; r < 8; ++r) {
    float v = c[r] + bv;
    if (act) v = gelu_t(v);
    tile[(r + 8 * hb) * ldw + col] = (__bf16)v;
  }
}

__device__ __forceinline__ float red16(float v) {
  v += __shfl_xor(v, 1, 32);
  v += __shfl_xor(v, 2, 32);
  v += __shfl_xor(v, 4, 32);
  v += __shfl_xor(v, 8, 32);
  return v;
}

__device__ __forceinline__ float norm_hash(unsigned idx) {
  unsigned x = idx * 2654435761u + 0x9E3779B9u;
  x ^= x >> 16; x *= 0x85EBCA6Bu; x ^= x >> 13; x *= 0xC2B2AE35u; x ^= x >> 16;
  unsigned y = (idx ^ 0xDEADBEEFu) * 2246822519u + 0x165667B1u;
  y ^= y >> 15; y *= 0x27D4EB2Fu; y ^= y >> 13;
  float u1 = (float)((x >> 8) + 1u) * (1.0f / 16777217.0f);
  float u2 = (float)(y >> 8) * (1.0f / 16777216.0f);
  return sqrtf(-2.0f * logf(u1)) * cosf(6.28318530718f * u2);
}

// ---------------------------------------------------------------------------
// Weight prep: f32 [K x N] row-major  ->  bf16 WMMA B-fragment layout
// ---------------------------------------------------------------------------
__global__ void k_prep(const float* __restrict__ W, unsigned short* __restrict__ dst,
                       int Kd, int Nd) {
  int idx = blockIdx.x * blockDim.x + threadIdx.x;
  if (idx >= Kd * Nd) return;
  int k = idx / Nd, n = idx % Nd;
  int NT = Nd >> 4;
  int kc = k >> 5, kr = k & 31;
  int hb = (kr >> 3) & 1, i = (kr & 7) + ((kr >> 4) << 3);
  int nt = n >> 4, lane = hb * 16 + (n & 15);
  __bf16 v = (__bf16)W[(size_t)k * Nd + n];
  ((__bf16*)dst)[(((size_t)(kc * NT + nt) * 32 + lane) << 4) + i] = v;
}

// ---------------------------------------------------------------------------
// Accumulator-resident inner layer over an LDS-staged 16x128 activation tile.
// kc loop kept dynamic (#pragma unroll 1) to bound in-flight B fragments.
// ---------------------------------------------------------------------------
__device__ __forceinline__ void layer_lds_128(__bf16* Ht, const __bf16* Wf,
                                              const float* bias, int lane,
                                              bool act, v8f acc_out[8]) {
  v8f acc[8];
#pragma unroll
  for (int nt = 0; nt < 8; ++nt) acc[nt] = (v8f){};
#pragma unroll 1
  for (int kc = 0; kc < 4; ++kc) {
    v16bf a = afrag_lds(Ht, DM, kc, lane);
#pragma unroll
    for (int nt = 0; nt < 8; ++nt)
      acc[nt] = wmma_bf16(a, bfrag(Wf, 8, kc, nt, lane), acc[nt]);
  }
  if (acc_out) {
#pragma unroll
    for (int nt = 0; nt < 8; ++nt) acc_out[nt] = acc[nt];
  } else {
#pragma unroll
    for (int nt = 0; nt < 8; ++nt) store_h(Ht, DM, nt, lane, acc[nt], bias, act);
  }
}

// ---------------------------------------------------------------------------
// seq stage: V = LN(V + MLP(concat(V,S)))   (256 -> 128 -> 128 -> 128)
// block=256 (8 waves), wave = 16 nodes. Per-wave LDS: 8KB (bf16 H / f32 tile)
// ---------------------------------------------------------------------------
__global__ void __launch_bounds__(256, 1)
k_seq(const float* __restrict__ V, const float* __restrict__ S,
      const unsigned short* W0u, const float* b0,
      const unsigned short* W1u, const float* b1,
      const unsigned short* W2u, const float* b2,
      const float* g, const float* be, float* __restrict__ Vout) {
  extern __shared__ char smem[];
  const int lane = threadIdx.x & 31, wave = threadIdx.x >> 5;
  char* wb = smem + wave * 8192;
  __bf16* Ht = (__bf16*)wb;
  float*  Ft = (float*)wb;
  const __bf16 *W0f = (const __bf16*)W0u, *W1f = (const __bf16*)W1u,
               *W2f = (const __bf16*)W2u;
  const int row0 = (blockIdx.x * 8 + wave) * 16;
  const int myrow = row0 + (lane & 15);
  const int hb = (lane >> 4) & 1, col0 = lane & 15;

  // layer1: stream 8 input K-chunks (4 from V, 4 from S), 8 accs live
  v8f acc[8];
#pragma unroll
  for (int nt = 0; nt < 8; ++nt) acc[nt] = (v8f){};
#pragma unroll 1
  for (int kc = 0; kc < 8; ++kc) {
    const float* src = (kc < 4) ? (V + (size_t)myrow * DM + kc * 32)
                                : (S + (size_t)myrow * DM + (kc - 4) * 32);
    v16bf a = afrag_g32(src, lane);
#pragma unroll
    for (int nt = 0; nt < 8; ++nt)
      acc[nt] = wmma_bf16(a, bfrag(W0f, 8, kc, nt, lane), acc[nt]);
  }
#pragma unroll
  for (int nt = 0; nt < 8; ++nt) store_h(Ht, DM, nt, lane, acc[nt], b0, true);

  layer_lds_128(Ht, W1f, b1, lane, true, nullptr);
  layer_lds_128(Ht, W2f, nullptr, lane, false, acc);

#pragma unroll
  for (int nt = 0; nt < 8; ++nt) {
    int col = nt * 16 + col0;
    float bv = b2[col];
#pragma unroll
    for (int r = 0; r < 8; ++r) {
      int row = r + 8 * hb;
      Ft[row * DM + col] = V[(size_t)(row0 + row) * DM + col] + acc[nt][r] + bv;
    }
  }
  if (lane < 16) {
    float s = 0.f, s2 = 0.f;
    for (int c = 0; c < DM; ++c) { float v = Ft[lane * DM + c]; s += v; s2 += v * v; }
    float mean = s * (1.f / DM), var = s2 * (1.f / DM) - mean * mean;
    float rs = rsqrtf(var + EPSL);
    for (int c = 0; c < DM; ++c)
      Vout[(size_t)(row0 + lane) * DM + c] =
          (Ft[lane * DM + c] - mean) * rs * g[c] + be[c];
  }
}

// ---------------------------------------------------------------------------
// msg stage: wave = one node (32 edges, 2 row tiles). Masked mean of
// MLP(concat(Vi,Vj,E)) with fused LN1:  Vb = LN(Va + dV)
// block=256 (8 waves = 8 nodes). Per-wave LDS: 4KB bf16 hidden tile.
// ---------------------------------------------------------------------------
__global__ void __launch_bounds__(256, 1)
k_msg(const float* __restrict__ Va, float* __restrict__ Vb,
      const float* __restrict__ Esrc, const int* __restrict__ Kd,
      const unsigned char* __restrict__ mask,
      const unsigned short* W0u, const float* b0,
      const unsigned short* W1u, const float* b1,
      const unsigned short* W2u, const float* b2,
      const float* ln_g, const float* ln_b) {
  extern __shared__ char smem[];
  const int lane = threadIdx.x & 31, wave = threadIdx.x >> 5;
  __bf16* Ht = (__bf16*)smem + wave * 16 * DM;
  const __bf16 *W0f = (const __bf16*)W0u, *W1f = (const __bf16*)W1u,
               *W2f = (const __bf16*)W2u;
  const int node  = blockIdx.x * 8 + wave;     // 0..8191
  const int batch = node >> 12;                // node / 4096
  const int e_base = node * KNB;
  const int hb = (lane >> 4) & 1, col0 = lane & 15;

  float partial[8];
#pragma unroll
  for (int nt = 0; nt < 8; ++nt) partial[nt] = 0.f;
  float cntp = 0.f;

#pragma unroll 1
  for (int t = 0; t < 2; ++t) {
    const int e_row = e_base + t * 16 + (lane & 15);  // A-matrix row edge
    const int j = batch * NN + Kd[e_row];

    // layer1: stream 12 input K-chunks (Vi, Vj, E), 8 accs live
    v8f acc[8];
#pragma unroll
    for (int nt = 0; nt < 8; ++nt) acc[nt] = (v8f){};
#pragma unroll 1
    for (int kc = 0; kc < 12; ++kc) {
      const float* src;
      if (kc < 4)       src = Va   + (size_t)node  * DM + kc * 32;
      else if (kc < 8)  src = Va   + (size_t)j     * DM + (kc - 4) * 32;
      else              src = Esrc + (size_t)e_row * DM + (kc - 8) * 32;
      v16bf a = afrag_g32(src, lane);
#pragma unroll
      for (int nt = 0; nt < 8; ++nt)
        acc[nt] = wmma_bf16(a, bfrag(W0f, 8, kc, nt, lane), acc[nt]);
    }
#pragma unroll
    for (int nt = 0; nt < 8; ++nt) store_h(Ht, DM, nt, lane, acc[nt], b0, true);

    layer_lds_128(Ht, W1f, b1, lane, true, nullptr);
    layer_lds_128(Ht, W2f, nullptr, lane, false, acc);

    float m[8];
#pragma unroll
    for (int r = 0; r < 8; ++r) {
      int e = e_base + t * 16 + r + 8 * hb;   // C-tile row edge
      m[r] = mask[e] ? 1.f : 0.f;
      cntp += m[r];
    }
#pragma unroll
    for (int nt = 0; nt < 8; ++nt) {
      float bv = b2[nt * 16 + col0];
#pragma unroll
      for (int r = 0; r < 8; ++r) partial[nt] += (acc[nt][r] + bv) * m[r];
    }
  }

  float cnt = cntp + __shfl_xor(cntp, 16, 32);
  float inv = 1.f / fmaxf(cnt, 1.f);
  float u[8], s = 0.f, s2 = 0.f;
#pragma unroll
  for (int nt = 0; nt < 8; ++nt) {
    float p = partial[nt] + __shfl_xor(partial[nt], 16, 32);
    float uu = Va[(size_t)node * DM + nt * 16 + col0] + p * inv;
    u[nt] = uu; s += uu; s2 += uu * uu;
  }
  s = red16(s); s2 = red16(s2);
  float mean = s * (1.f / DM), var = s2 * (1.f / DM) - mean * mean;
  float rs = rsqrtf(var + EPSL);
  if (lane < 16) {
#pragma unroll
    for (int nt = 0; nt < 8; ++nt) {
      int col = nt * 16 + col0;
      Vb[(size_t)node * DM + col] = (u[nt] - mean) * rs * ln_g[col] + ln_b[col];
    }
  }
}

// ---------------------------------------------------------------------------
// node stage: Va = LN2(Vb + FFN(Vb))  (128 -> 512 gelu -> 128)
// block=128 (4 waves), wave = 16 nodes. Per-wave LDS: 16KB bf16 hidden tile
// (f32 residual tile overlays the first 8KB after the 2nd GEMM's LDS reads).
// ---------------------------------------------------------------------------
__global__ void __launch_bounds__(128, 1)
k_node(const float* __restrict__ Vb, float* __restrict__ Va,
       const unsigned short* W0u, const float* b0,
       const unsigned short* W1u, const float* b1,
       const float* g, const float* be) {
  extern __shared__ char smem[];
  const int lane = threadIdx.x & 31, wave = threadIdx.x >> 5;
  char* wb = smem + wave * 16384;
  __bf16* Hb = (__bf16*)wb;    // 16 x 512 bf16
  float*  Ft = (float*)wb;     // 16 x 128 f32 overlay
  const __bf16 *W0f = (const __bf16*)W0u, *W1f = (const __bf16*)W1u;
  const int n0 = (blockIdx.x * 4 + wave) * 16;
  const int hb = (lane >> 4) & 1, col0 = lane & 15;

  v16bf av[4];
#pragma unroll
  for (int kc = 0; kc < 4; ++kc)
    av[kc] = afrag_g32(Vb + (size_t)(n0 + (lane & 15)) * DM + kc * 32, lane);

#pragma unroll 1
  for (int nt = 0; nt < 32; ++nt) {
    v8f c = {};
#pragma unroll
    for (int kc = 0; kc < 4; ++kc)
      c = wmma_bf16(av[kc], bfrag(W0f, 32, kc, nt, lane), c);
    store_h(Hb, 512, nt, lane, c, b0, true);
  }

  // 2nd GEMM: stream 16 hidden K-chunks from LDS, 8 accs live
  v8f acc[8];
#pragma unroll
  for (int nt = 0; nt < 8; ++nt) acc[nt] = (v8f){};
#pragma unroll 1
  for (int kc = 0; kc < 16; ++kc) {
    v16bf hk = afrag_lds(Hb, 512, kc, lane);
#pragma unroll
    for (int nt = 0; nt < 8; ++nt)
      acc[nt] = wmma_bf16(hk, bfrag(W1f, 8, kc, nt, lane), acc[nt]);
  }
#pragma unroll
  for (int nt = 0; nt < 8; ++nt) {
    int col = nt * 16 + col0;
    float bv = b1[col];
#pragma unroll
    for (int r = 0; r < 8; ++r) {
      int row = r + 8 * hb;
      Ft[row * DM + col] = Vb[(size_t)(n0 + row) * DM + col] + acc[nt][r] + bv;
    }
  }
  if (lane < 16) {
    float s = 0.f, s2 = 0.f;
    for (int c = 0; c < DM; ++c) { float v = Ft[lane * DM + c]; s += v; s2 += v * v; }
    float mean = s * (1.f / DM), var = s2 * (1.f / DM) - mean * mean;
    float rs = rsqrtf(var + EPSL);
    for (int c = 0; c < DM; ++c)
      Va[(size_t)(n0 + lane) * DM + c] =
          (Ft[lane * DM + c] - mean) * rs * g[c] + be[c];
  }
}

// ---------------------------------------------------------------------------
// edge stage: Edst = LN(Esrc + MLP(concat(Vi2,Vj2,E)))
// block=256 (8 waves), wave = 16 edges. Per-wave LDS: 8KB overlay.
// ---------------------------------------------------------------------------
__global__ void __launch_bounds__(256, 1)
k_edge(const float* __restrict__ Va, const float* __restrict__ Esrc,
       float* __restrict__ Edst, const int* __restrict__ Kd,
       const unsigned short* W0u, const float* b0,
       const unsigned short* W1u, const float* b1,
       const unsigned short* W2u, const float* b2,
       const float* g, const float* be) {
  extern __shared__ char smem[];
  const int lane = threadIdx.x & 31, wave = threadIdx.x >> 5;
  char* wb = smem + wave * 8192;
  __bf16* Ht = (__bf16*)wb;
  float*  Ft = (float*)wb;
  const __bf16 *W0f = (const __bf16*)W0u, *W1f = (const __bf16*)W1u,
               *W2f = (const __bf16*)W2u;
  const int e0 = (blockIdx.x * 8 + wave) * 16;
  const int e_row = e0 + (lane & 15);
  const int n_g = e_row >> 5;          // global node (includes batch)
  const int batch = e_row >> 17;       // e / (N*Kn)
  const int j = batch * NN + Kd[e_row];
  const int hb = (lane >> 4) & 1, col0 = lane & 15;

  // layer1: stream 12 K-chunks, 8 accs live
  v8f acc[8];
#pragma unroll
  for (int nt = 0; nt < 8; ++nt) acc[nt] = (v8f){};
#pragma unroll 1
  for (int kc = 0; kc < 12; ++kc) {
    const float* src;
    if (kc < 4)       src = Va   + (size_t)n_g   * DM + kc * 32;
    else if (kc < 8)  src = Va   + (size_t)j     * DM + (kc - 4) * 32;
    else              src = Esrc + (size_t)e_row * DM + (kc - 8) * 32;
    v16bf a = afrag_g32(src, lane);
#pragma unroll
    for (int nt = 0; nt < 8; ++nt)
      acc[nt] = wmma_bf16(a, bfrag(W0f, 8, kc, nt, lane), acc[nt]);
  }
#pragma unroll
  for (int nt = 0; nt < 8; ++nt) store_h(Ht, DM, nt, lane, acc[nt], b0, true);

  layer_lds_128(Ht, W1f, b1, lane, true, nullptr);
  layer_lds_128(Ht, W2f, nullptr, lane, false, acc);

#pragma unroll
  for (int nt = 0; nt < 8; ++nt) {
    int col = nt * 16 + col0;
    float bv = b2[col];
#pragma unroll
    for (int r = 0; r < 8; ++r) {
      int e = e0 + r + 8 * hb;
      Ft[(r + 8 * hb) * DM + col] = Esrc[(size_t)e * DM + col] + acc[nt][r] + bv;
    }
  }
  if (lane < 16) {
    int e = e0 + lane;
    float s = 0.f, s2 = 0.f;
    for (int c = 0; c < DM; ++c) { float v = Ft[lane * DM + c]; s += v; s2 += v * v; }
    float mean = s * (1.f / DM), var = s2 * (1.f / DM) - mean * mean;
    float rs = rsqrtf(var + EPSL);
    for (int c = 0; c < DM; ++c)
      Edst[(size_t)e * DM + c] = (Ft[lane * DM + c] - mean) * rs * g[c] + be[c];
  }
}

// ---------------------------------------------------------------------------
// latent stage: mu,logvar = MLP(V) (128 -> 512 gelu -> 256), Z = mu + n*exp(.5lv)
// block=128 (4 waves), wave = 16 nodes.
// ---------------------------------------------------------------------------
__global__ void __launch_bounds__(128, 1)
k_latent(const float* __restrict__ Va,
         const unsigned short* W0u, const float* b0,
         const unsigned short* W1u, const float* b1,
         float* __restrict__ out, unsigned bnd) {
  extern __shared__ char smem[];
  const int lane = threadIdx.x & 31, wave = threadIdx.x >> 5;
  char* wb = smem + wave * 16384;
  __bf16* Hb = (__bf16*)wb;
  const __bf16 *W0f = (const __bf16*)W0u, *W1f = (const __bf16*)W1u;
  const int n0 = (blockIdx.x * 4 + wave) * 16;
  const int hb = (lane >> 4) & 1, col0 = lane & 15;

  v16bf av[4];
#pragma unroll
  for (int kc = 0; kc < 4; ++kc)
    av[kc] = afrag_g32(Va + (size_t)(n0 + (lane & 15)) * DM + kc * 32, lane);

#pragma unroll 1
  for (int nt = 0; nt < 32; ++nt) {
    v8f c = {};
#pragma unroll
    for (int kc = 0; kc < 4; ++kc)
      c = wmma_bf16(av[kc], bfrag(W0f, 32, kc, nt, lane), c);
    store_h(Hb, 512, nt, lane, c, b0, true);
  }

  // per output pair (mu tile nt, logvar tile nt+8): only 2 accs live,
  // hidden chunks streamed from LDS.
#pragma unroll 1
  for (int nt = 0; nt < 8; ++nt) {
    v8f cm = {}, cl = {};
#pragma unroll 1
    for (int kc = 0; kc < 16; ++kc) {
      v16bf hk = afrag_lds(Hb, 512, kc, lane);
      cm = wmma_bf16(hk, bfrag(W1f, 16, kc, nt, lane), cm);
      cl = wmma_bf16(hk, bfrag(W1f, 16, kc, nt + 8, lane), cl);
    }
    int col = nt * 16 + col0;
    float bm = b1[col], bl = b1[col + DM];
#pragma unroll
    for (int r = 0; r < 8; ++r) {
      int row = n0 + r + 8 * hb;
      float mu = cm[r] + bm, lv = cl[r] + bl;
      size_t idx = (size_t)row * DM + col;
      float nz = norm_hash((unsigned)idx);
      out[idx]                   = mu + nz * expf(0.5f * lv);  // Z
      out[bnd + idx]             = mu;                         // Z_mu
      out[2 * (size_t)bnd + idx] = lv;                         // Z_logvar
    }
  }
}

// ---------------------------------------------------------------------------
// Host side
// ---------------------------------------------------------------------------
extern "C" void kernel_launch(void* const* d_in, const int* in_sizes, int n_in,
                              void* d_out, int out_size, void* d_ws, size_t ws_size,
                              hipStream_t stream) {
  const size_t BND = (size_t)BNN * DM;      // 1,048,576
  char* ws = (char*)d_ws;
  size_t off = 0;
  auto carve = [&](size_t bytes) -> char* {
    char* p = ws + off;
    off = (off + bytes + 255) & ~(size_t)255;
    return p;
  };
  float* Va   = (float*)carve(BND * 4);
  float* Vb   = (float*)carve(BND * 4);
  float* Ebuf = (float*)carve((size_t)BNN * KNB * DM * 4);
  auto carve_w = [&](int K, int N) { return (unsigned short*)carve((size_t)K * N * 2); };

  unsigned short* wSeq[3]  = { carve_w(256, 128), carve_w(128, 128), carve_w(128, 128) };
  unsigned short* wMsg[3][3];
  unsigned short* wEdge[3][3];
  unsigned short* wFfn[3][2];
  for (int l = 0; l < 3; ++l) {
    wEdge[l][0] = carve_w(384, 128); wEdge[l][1] = carve_w(128, 128); wEdge[l][2] = carve_w(128, 128);
    wFfn[l][0]  = carve_w(128, 512); wFfn[l][1]  = carve_w(512, 128);
    wMsg[l][0]  = carve_w(384, 128); wMsg[l][1]  = carve_w(128, 128); wMsg[l][2]  = carve_w(128, 128);
  }
  unsigned short* wLat0 = carve_w(128, 512);
  unsigned short* wLat1 = carve_w(512, 256);

  // Input indices (jax pytree: dicts flattened with sorted keys)
  //  0:V 1:E 2:K 3:S 4:edge_mask
  //  latent: 5:W0 6:b0 7:W1 8:b1
  //  layers[l] base = 9 + 22*l :
  //    +0..5  edge (W,b)x3 ; +6,7 edge_ln(g,b); +8..11 ffn (W,b)x2
  //    +12,13 ln1 ; +14,15 ln2 ; +16..21 msg (W,b)x3
  //  75:seq_ln g 76:seq_ln b ; 77..82 seq_msg (W,b)x3
  auto F  = [&](int i) { return (const float*)d_in[i]; };
  auto prep = [&](int idx, unsigned short* dst, int K, int N) {
    int total = K * N;
    k_prep<<<(total + 255) / 256, 256, 0, stream>>>(F(idx), dst, K, N);
  };

  prep(5, wLat0, 128, 512);
  prep(7, wLat1, 512, 256);
  prep(77, wSeq[0], 256, 128);
  prep(79, wSeq[1], 128, 128);
  prep(81, wSeq[2], 128, 128);
  for (int l = 0; l < 3; ++l) {
    int b = 9 + 22 * l;
    prep(b + 0,  wEdge[l][0], 384, 128);
    prep(b + 2,  wEdge[l][1], 128, 128);
    prep(b + 4,  wEdge[l][2], 128, 128);
    prep(b + 8,  wFfn[l][0],  128, 512);
    prep(b + 10, wFfn[l][1],  512, 128);
    prep(b + 16, wMsg[l][0],  384, 128);
    prep(b + 18, wMsg[l][1],  128, 128);
    prep(b + 20, wMsg[l][2],  128, 128);
  }

  const int* Kd = (const int*)d_in[2];
  const unsigned char* mask = (const unsigned char*)d_in[4];

  // seq: V,S -> Va
  k_seq<<<BNN / 128, 256, 8 * 8192, stream>>>(F(0), F(3),
      wSeq[0], F(78), wSeq[1], F(80), wSeq[2], F(82), F(75), F(76), Va);

  for (int l = 0; l < 3; ++l) {
    int b = 9 + 22 * l;
    const float* Esrc = (l == 0) ? F(1) : Ebuf;
    // msg: Va -> Vb (fused LN1)
    k_msg<<<BNN / 8, 256, 8 * 4096, stream>>>(Va, Vb, Esrc, Kd, mask,
        wMsg[l][0], F(b + 17), wMsg[l][1], F(b + 19), wMsg[l][2], F(b + 21),
        F(b + 12), F(b + 13));
    // node: Vb -> Va (FFN + LN2)
    k_node<<<BNN / 64, 128, 4 * 16384, stream>>>(Vb, Va,
        wFfn[l][0], F(b + 9), wFfn[l][1], F(b + 11), F(b + 14), F(b + 15));
    // edge: (Va, Esrc) -> Ebuf (fused edge LN)
    k_edge<<<(BNN * KNB) / 128, 256, 8 * 8192, stream>>>(Va, Esrc, Ebuf, Kd,
        wEdge[l][0], F(b + 1), wEdge[l][1], F(b + 3), wEdge[l][2], F(b + 5),
        F(b + 6), F(b + 7));
  }

  // latent: Va -> (Z, mu, logvar)
  k_latent<<<BNN / 64, 128, 4 * 16384, stream>>>(Va,
      wLat0, F(6), wLat1, F(8), (float*)d_out, (unsigned)BND);

  (void)in_sizes; (void)n_in; (void)out_size; (void)ws_size;
}
